// MultiHeadAttention_12412455485842
// MI455X (gfx1250) — matrix operand order repaired and checked by reading, not compiled, not run on hardware
//
#include <hip/hip_runtime.h>
#include <hip/hip_bf16.h>

// ---------------------------------------------------------------------------
// MHA forward: B=4, S=2048, D=1024, H=16, Hd=64, causal, scale = 1/8
// Pipeline:
//   0) fp32 -> bf16 pre-convert of x and W_q/k/v/o (bandwidth pass, done once)
//   1) QKV projection  : bf16 WMMA GEMM, 256x64 tile, K-stage 64
//   2) flash attention : 64-key chunks, online softmax, all matmuls WMMA
//   3) output projection: bf16 WMMA GEMM -> fp32 out
// ---------------------------------------------------------------------------

#define Bx   4
#define Sx   2048
#define Dx   1024
#define Hx   16
#define Hdx  64
#define ROWS (Bx * Sx)         // 8192
#define SCALE_INV 0.125f

typedef __attribute__((ext_vector_type(16))) __bf16 v16bf;
typedef __attribute__((ext_vector_type(8)))  float  v8f;

__device__ __forceinline__ v8f wmma_bf16(v16bf a, v16bf b, v8f c) {
    return __builtin_amdgcn_wmma_f32_16x16x32_bf16(false, a, false, b,
                                                   (short)0, c, false, false);
}

// A-matrix 16x32 bf16 fragment (ISA 7.12.2): lane = M (0..15), half = lane>>4.
// VGPR p<4: K = 2p,2p+1 (+8*half); p>=4: K = 16+2(p-4) (+8*half).
__device__ __forceinline__ v16bf load_frag_a(const __bf16* rowptr, int hf) {
    union { v16bf v; unsigned int u[8]; } f;
#pragma unroll
    for (int p = 0; p < 8; ++p) {
        const int k0 = ((p < 4) ? (2 * p) : (16 + 2 * (p - 4))) + 8 * hf;
        f.u[p] = *reinterpret_cast<const unsigned int*>(rowptr + k0);
    }
    return f.v;
}

// B-matrix 32x16 bf16 fragment: lane = N (0..15), half = lane>>4.
// VGPR p<4: K = 2p,2p+1 (+8*half); p>=4: K = 16+2(p-4) (+8*half). rowptr = lane's B-column, contiguous in K.
__device__ __forceinline__ v16bf load_frag_b(const __bf16* rowptr, int hf) {
    union { v16bf v; unsigned int u[8]; } f;
#pragma unroll
    for (int p = 0; p < 8; ++p) {
        const int k0 = ((p < 4) ? (2 * p) : (16 + 2 * (p - 4))) + 8 * hf;
        f.u[p] = *reinterpret_cast<const unsigned int*>(rowptr + k0);
    }
    return f.v;
}

__device__ __forceinline__ float rmax16(float v) {
#pragma unroll
    for (int m = 1; m <= 8; m <<= 1) v = fmaxf(v, __shfl_xor(v, m, 32));
    return v;
}
__device__ __forceinline__ float rsum16(float v) {
#pragma unroll
    for (int m = 1; m <= 8; m <<= 1) v += __shfl_xor(v, m, 32);
    return v;
}

// ---------------------------------------------------------------------------
// Kernel 0: fp32 -> bf16 (8 elements/thread, uint4 stores).
// ---------------------------------------------------------------------------
__global__ void __launch_bounds__(256)
cvt_bf16_kernel(const float* __restrict__ src, __bf16* __restrict__ dst, int n8) {
    int i = blockIdx.x * blockDim.x + threadIdx.x;
    if (i >= n8) return;
    const float4* s = reinterpret_cast<const float4*>(src);
    float4 a = s[2 * i], b = s[2 * i + 1];
    union { __bf16 h[8]; uint4 u; } r;
    r.h[0] = (__bf16)a.x; r.h[1] = (__bf16)a.y;
    r.h[2] = (__bf16)a.z; r.h[3] = (__bf16)a.w;
    r.h[4] = (__bf16)b.x; r.h[5] = (__bf16)b.y;
    r.h[6] = (__bf16)b.z; r.h[7] = (__bf16)b.w;
    reinterpret_cast<uint4*>(dst)[i] = r.u;
}

// ---------------------------------------------------------------------------
// Kernel 1: QKV projection. C[g][n] = sum_k xb[g][k] * Wb[n][k]   (bf16 WMMA)
// 256x64 tile / block (8 waves x 32 rows), K-stage 64. grid=(32, 16, 3).
// ---------------------------------------------------------------------------
__global__ void __launch_bounds__(256)
qkv_proj_kernel(const __bf16* __restrict__ xb,
                const __bf16* __restrict__ Wqb,
                const __bf16* __restrict__ Wkb,
                const __bf16* __restrict__ Wvb,
                __bf16* __restrict__ qb,
                __bf16* __restrict__ kb,
                __bf16* __restrict__ vb) {
    __shared__ __align__(16) __bf16 As[256 * 64];   // 32 KB
    __shared__ __align__(16) __bf16 Bs[64 * 64];    //  8 KB

    const int tid  = threadIdx.x;
    const int wave = tid >> 5;
    const int lane = tid & 31;
    const int hf   = lane >> 4;
    const int lr   = lane & 15;

    const int rowBase = blockIdx.x * 256;
    const int nBase   = blockIdx.y * 64;
    const int z       = blockIdx.z;
    const __bf16* W   = (z == 0) ? Wqb : (z == 1) ? Wkb : Wvb;
    __bf16*       dst = (z == 0) ? qb  : (z == 1) ? kb  : vb;

    const uint4* xs = reinterpret_cast<const uint4*>(xb);  // row stride 128 u4
    const uint4* ws = reinterpret_cast<const uint4*>(W);

    v8f acc0[4] = {}, acc1[4] = {};

    for (int kk = 0; kk < Dx; kk += 64) {
#pragma unroll
        for (int i = 0; i < 8; ++i) {    // As: 2048 uint4
            int e = tid + i * 256; int r = e >> 3, q = e & 7;
            reinterpret_cast<uint4*>(As)[e] =
                xs[(size_t)(rowBase + r) * 128 + (kk >> 3) + q];
        }
#pragma unroll
        for (int i = 0; i < 2; ++i) {    // Bs: 512 uint4
            int e = tid + i * 256; int r = e >> 3, q = e & 7;
            reinterpret_cast<uint4*>(Bs)[e] =
                ws[(size_t)(nBase + r) * 128 + (kk >> 3) + q];
        }
        if (kk + 64 < Dx)               // gfx1250 global_prefetch_b8
            __builtin_prefetch(xs + (size_t)(rowBase + tid) * 128 + ((kk + 64) >> 3), 0, 0);
        __syncthreads();

#pragma unroll
        for (int ks = 0; ks < 64; ks += 32) {
            v16bf af0 = load_frag_a(&As[(wave * 32 + lr) * 64 + ks], hf);
            v16bf af1 = load_frag_a(&As[(wave * 32 + 16 + lr) * 64 + ks], hf);
#pragma unroll
            for (int nb = 0; nb < 4; ++nb) {
                v16bf bfr = load_frag_b(&Bs[(nb * 16 + lr) * 64 + ks], hf);
                acc0[nb] = wmma_bf16(af0, bfr, acc0[nb]);
                acc1[nb] = wmma_bf16(af1, bfr, acc1[nb]);
            }
        }
        __syncthreads();
    }

#pragma unroll
    for (int grp = 0; grp < 2; ++grp)
#pragma unroll
        for (int nb = 0; nb < 4; ++nb)
#pragma unroll
            for (int j = 0; j < 8; ++j) {
                float v = grp ? acc1[nb][j] : acc0[nb][j];
                int g = rowBase + wave * 32 + grp * 16 + j + 8 * hf;
                int n = nBase + nb * 16 + lr;
                int b = g >> 11, s = g & (Sx - 1);
                int h = n >> 6, d = n & (Hdx - 1);
                dst[(((size_t)b * Hx + h) * Sx + s) * Hdx + d] = (__bf16)v;
            }
}

// ---------------------------------------------------------------------------
// Kernel 2: causal flash attention. grid = (S/128, B*H), block = 256 (8 waves).
// Each wave owns a 16-query tile; key chunks of 64 staged in LDS.
// ---------------------------------------------------------------------------
__global__ void __launch_bounds__(256)
attn_kernel(const __bf16* __restrict__ Qb,
            const __bf16* __restrict__ Kb,
            const __bf16* __restrict__ Vb,
            __bf16* __restrict__ AO) {
    __shared__ __align__(16) __bf16 Ks[64 * 64];      // [key][d]      8 KB
    __shared__ __align__(16) __bf16 Vt[64 * 64];      // [d][key]      8 KB
    __shared__ __align__(16) __bf16 Pst[8 * 16 * 64]; // per-wave P   16 KB

    const int tid  = threadIdx.x;
    const int wave = tid >> 5;
    const int lane = tid & 31;
    const int hf   = lane >> 4;
    const int lr   = lane & 15;

    const int bh    = blockIdx.y;               // b*H + h
    const int qbase = blockIdx.x * 128 + wave * 16;

    const __bf16* Qg = Qb + (size_t)bh * Sx * Hdx;
    const __bf16* Kg = Kb + (size_t)bh * Sx * Hdx;
    const __bf16* Vg = Vb + (size_t)bh * Sx * Hdx;
    __bf16* Pw = &Pst[wave * 16 * 64];

    // Q fragments (16 rows x 64 d), loaded once from global.
    const __bf16* qrow = Qg + (size_t)(qbase + lr) * Hdx;
    v16bf qf0 = load_frag_a(qrow, hf);
    v16bf qf1 = load_frag_a(qrow + 32, hf);

    v8f   accs[4] = {};
    float m[8], lsum[8];
#pragma unroll
    for (int j = 0; j < 8; ++j) { m[j] = -1e30f; lsum[j] = 0.0f; }

    const int nchunks = (blockIdx.x + 1) * 2;   // 64-key chunks, causal limit
    for (int ck = 0; ck < nchunks; ++ck) {
        const int kbase = ck * 64;

        // cooperative stage: K chunk contiguous copy, V chunk transposed
        {
            const uint4* ksrc = reinterpret_cast<const uint4*>(Kg) + (size_t)kbase * 8;
#pragma unroll
            for (int i = 0; i < 2; ++i)
                reinterpret_cast<uint4*>(Ks)[tid + i * 256] = ksrc[tid + i * 256];
#pragma unroll
            for (int i = 0; i < 16; ++i) {
                int e = tid + i * 256; int key = e >> 6; int d = e & 63;
                Vt[d * 64 + key] = Vg[(size_t)(kbase + key) * Hdx + d];
            }
        }
        __syncthreads();

        // scores: four 16x16 tiles, contraction over Hd=64 (2 WMMAs each)
        v8f sa[4];
#pragma unroll
        for (int t = 0; t < 4; ++t) {
            const __bf16* kr = &Ks[(t * 16 + lr) * 64];
            v8f s = {};
            s = wmma_bf16(qf0, load_frag_b(kr, hf), s);
            s = wmma_bf16(qf1, load_frag_b(kr + 32, hf), s);
            sa[t] = s;
        }

#pragma unroll
        for (int j = 0; j < 8; ++j) {
            const int qg = qbase + j + 8 * hf;
            float sv[4]; bool ok[4];
            float tmax = -1e30f;
#pragma unroll
            for (int t = 0; t < 4; ++t) {
                ok[t] = (kbase + t * 16 + lr) <= qg;
                sv[t] = ok[t] ? sa[t][j] * SCALE_INV : -1e30f;
                tmax = fmaxf(tmax, sv[t]);
            }
            float mnew = fmaxf(m[j], rmax16(tmax));
            float psum = 0.0f;
#pragma unroll
            for (int t = 0; t < 4; ++t) {
                float p = ok[t] ? __expf(sv[t] - mnew) : 0.0f;
                psum += p;
                Pw[(j + 8 * hf) * 64 + t * 16 + lr] = (__bf16)p;
            }
            float alpha = __expf(m[j] - mnew);
            lsum[j] = lsum[j] * alpha + rsum16(psum);
            m[j] = mnew;
#pragma unroll
            for (int nb = 0; nb < 4; ++nb) accs[nb][j] *= alpha;
        }
        __syncthreads();   // P staging visible (C-layout -> A-layout round trip)

        // out += P (16x64) @ V (64x64): 2 K-substeps x 4 N-blocks
        v16bf pf0 = load_frag_a(&Pw[lr * 64], hf);
        v16bf pf1 = load_frag_a(&Pw[lr * 64 + 32], hf);
#pragma unroll
        for (int nb = 0; nb < 4; ++nb) {
            accs[nb] = wmma_bf16(pf0, load_frag_b(&Vt[(nb * 16 + lr) * 64], hf), accs[nb]);
            accs[nb] = wmma_bf16(pf1, load_frag_b(&Vt[(nb * 16 + lr) * 64 + 32], hf), accs[nb]);
        }
        __syncthreads();   // before next chunk overwrites Ks/Vt
    }

    // normalize + merge heads: AO[b, s, h*64 + d]  (bf16)
    const int b = bh >> 4, h = bh & 15;
#pragma unroll
    for (int j = 0; j < 8; ++j) {
        const float inv = 1.0f / lsum[j];
        const int s = qbase + j + 8 * hf;
#pragma unroll
        for (int nb = 0; nb < 4; ++nb) {
            AO[((size_t)b * Sx + s) * Dx + h * Hdx + nb * 16 + lr] =
                (__bf16)(accs[nb][j] * inv);
        }
    }
}

// ---------------------------------------------------------------------------
// Kernel 3: output projection. out[g][n] = sum_k AO[g][k] * Wob[n][k] (fp32)
// Same tiling as kernel 1. grid = (32, 16).
// ---------------------------------------------------------------------------
__global__ void __launch_bounds__(256)
out_proj_kernel(const __bf16* __restrict__ AO,
                const __bf16* __restrict__ Wob,
                float* __restrict__ out) {
    __shared__ __align__(16) __bf16 As[256 * 64];
    __shared__ __align__(16) __bf16 Bs[64 * 64];

    const int tid  = threadIdx.x;
    const int wave = tid >> 5;
    const int lane = tid & 31;
    const int hf   = lane >> 4;
    const int lr   = lane & 15;

    const int rowBase = blockIdx.x * 256;
    const int nBase   = blockIdx.y * 64;

    const uint4* xs = reinterpret_cast<const uint4*>(AO);
    const uint4* ws = reinterpret_cast<const uint4*>(Wob);

    v8f acc0[4] = {}, acc1[4] = {};

    for (int kk = 0; kk < Dx; kk += 64) {
#pragma unroll
        for (int i = 0; i < 8; ++i) {
            int e = tid + i * 256; int r = e >> 3, q = e & 7;
            reinterpret_cast<uint4*>(As)[e] =
                xs[(size_t)(rowBase + r) * 128 + (kk >> 3) + q];
        }
#pragma unroll
        for (int i = 0; i < 2; ++i) {
            int e = tid + i * 256; int r = e >> 3, q = e & 7;
            reinterpret_cast<uint4*>(Bs)[e] =
                ws[(size_t)(nBase + r) * 128 + (kk >> 3) + q];
        }
        if (kk + 64 < Dx)
            __builtin_prefetch(xs + (size_t)(rowBase + tid) * 128 + ((kk + 64) >> 3), 0, 0);
        __syncthreads();

#pragma unroll
        for (int ks = 0; ks < 64; ks += 32) {
            v16bf af0 = load_frag_a(&As[(wave * 32 + lr) * 64 + ks], hf);
            v16bf af1 = load_frag_a(&As[(wave * 32 + 16 + lr) * 64 + ks], hf);
#pragma unroll
            for (int nb = 0; nb < 4; ++nb) {
                v16bf bfr = load_frag_b(&Bs[(nb * 16 + lr) * 64 + ks], hf);
                acc0[nb] = wmma_bf16(af0, bfr, acc0[nb]);
                acc1[nb] = wmma_bf16(af1, bfr, acc1[nb]);
            }
        }
        __syncthreads();
    }

#pragma unroll
    for (int grp = 0; grp < 2; ++grp)
#pragma unroll
        for (int nb = 0; nb < 4; ++nb)
#pragma unroll
            for (int j = 0; j < 8; ++j) {
                int g = rowBase + wave * 32 + grp * 16 + j + 8 * hf;
                int n = nBase + nb * 16 + lr;
                out[(size_t)g * Dx + n] = grp ? acc1[nb][j] : acc0[nb][j];
            }
}

// ---------------------------------------------------------------------------
extern "C" void kernel_launch(void* const* d_in, const int* in_sizes, int n_in,
                              void* d_out, int out_size, void* d_ws, size_t ws_size,
                              hipStream_t stream) {
    (void)in_sizes; (void)n_in; (void)out_size; (void)ws_size;

    const float* x  = (const float*)d_in[0];
    const float* Wq = (const float*)d_in[1];
    const float* Wk = (const float*)d_in[2];
    const float* Wv = (const float*)d_in[3];
    const float* Wo = (const float*)d_in[4];
    float* out = (float*)d_out;

    // workspace (bf16): Q,K,V [B,H,S,Hd], AO [B,S,D], xb [B,S,D], 4 weights
    const size_t per   = (size_t)Bx * Hx * Sx * Hdx;   // 8,388,608
    const size_t wsz   = (size_t)Dx * Dx;              // 1,048,576
    __bf16* qb  = (__bf16*)d_ws;
    __bf16* kb  = qb + per;
    __bf16* vb  = kb + per;
    __bf16* ao  = vb + per;
    __bf16* xbf = ao + per;
    __bf16* wqb = xbf + per;
    __bf16* wkb = wqb + wsz;
    __bf16* wvb = wkb + wsz;
    __bf16* wob = wvb + wsz;

    dim3 blk(256);
    cvt_bf16_kernel<<<dim3((int)(per / 8 / 256)), blk, 0, stream>>>(x, xbf, (int)(per / 8));
    cvt_bf16_kernel<<<dim3((int)(wsz / 8 / 256)), blk, 0, stream>>>(Wq, wqb, (int)(wsz / 8));
    cvt_bf16_kernel<<<dim3((int)(wsz / 8 / 256)), blk, 0, stream>>>(Wk, wkb, (int)(wsz / 8));
    cvt_bf16_kernel<<<dim3((int)(wsz / 8 / 256)), blk, 0, stream>>>(Wv, wvb, (int)(wsz / 8));
    cvt_bf16_kernel<<<dim3((int)(wsz / 8 / 256)), blk, 0, stream>>>(Wo, wob, (int)(wsz / 8));

    qkv_proj_kernel<<<dim3(ROWS / 256, Dx / 64, 3), blk, 0, stream>>>(
        xbf, wqb, wkb, wvb, qb, kb, vb);
    attn_kernel<<<dim3(Sx / 128, Bx * Hx), blk, 0, stream>>>(qb, kb, vb, ao);
    out_proj_kernel<<<dim3(ROWS / 256, Dx / 64), blk, 0, stream>>>(ao, wob, out);
}
